// FlowTActor_31207232372995
// MI455X (gfx1250) — matrix operand order repaired and checked by reading, not compiled, not run on hardware
//
#include <hip/hip_runtime.h>
#include <hip/hip_bf16.h>

// ---------------------------------------------------------------------------
// FlowTActor forward on gfx1250: bf16 WMMA GEMM tower + wave32 elementwise.
// ---------------------------------------------------------------------------

#define B_ROWS 8192
#define DMODEL 512
#define ADIM   64
#define NSTEPS 4
#define NLAYER 4

typedef __attribute__((ext_vector_type(16))) __bf16 v16bf;
typedef __attribute__((ext_vector_type(8)))  __bf16 v8bf;
typedef __attribute__((ext_vector_type(8)))  float  v8f;

union FragAB { v16bf v; v8bf h[2]; };

__device__ __forceinline__ float dev_silu(float v) { return v / (1.f + __expf(-v)); }
__device__ __forceinline__ float dev_gelu(float v) { return 0.5f * v * (1.f + erff(v * 0.70710678118654752f)); }

// ---------------------------------------------------------------------------
// Weight repack: f32 row-major [K,N]  ->  bf16 WMMA-B fragment order.
// Tile = 32(K) x 16(N). dst element order: [nt][kt][lane][vgpr*2+half]
// lane<16: n=nt*16+lane,   k=kt*32 + q      (q = vgpr*2+half, 0..15)
// lane>=16: n=nt*16+lane-16, k=kt*32+16 + q
// => every lane's fragment is 32 contiguous bytes; k-tiles contiguous per nt.
// ---------------------------------------------------------------------------
__global__ __launch_bounds__(256) void pack_b_kernel(const float* __restrict__ src,
                                                     __bf16* __restrict__ dst,
                                                     int K, int N) {
  int idx = blockIdx.x * 256 + threadIdx.x;
  int total = K * N;
  if (idx >= total) return;
  int ktiles = K >> 5;
  int t    = idx >> 9;       // /512 elems per tile
  int r    = idx & 511;
  int lane = r >> 4;
  int q    = r & 15;
  int nt   = t / ktiles;
  int kt   = t - nt * ktiles;
  int n = nt * 16 + (lane & 15);
  int k = kt * 32 + ((lane >> 4) << 4) + q;
  dst[idx] = (__bf16)src[(size_t)k * N + n];
}

__global__ __launch_bounds__(256) void f32_to_bf16_kernel(const float* __restrict__ src,
                                                          __bf16* __restrict__ dst, int n) {
  int i = blockIdx.x * 256 + threadIdx.x;
  if (i < n) dst[i] = (__bf16)src[i];
}

__global__ __launch_bounds__(256) void x_init_kernel(const float* __restrict__ x0,
                                                     float* __restrict__ xf,
                                                     __bf16* __restrict__ xb, int n) {
  int i = blockIdx.x * 256 + threadIdx.x;
  if (i < n) { float v = x0[i]; xf[i] = v; xb[i] = (__bf16)v; }
}

__global__ __launch_bounds__(256) void lp_init_kernel(const float* __restrict__ x0,
                                                      float* __restrict__ lp) {
  int row = blockIdx.x * 256 + threadIdx.x;
  if (row >= B_ROWS) return;
  const float log2pi = 1.8378770664093453f;
  float s = 0.f;
  #pragma unroll 8
  for (int j = 0; j < ADIM; ++j) {
    float v = x0[(size_t)row * ADIM + j];
    s += -0.5f * v * v - 0.5f * log2pi;
  }
  lp[row] = s;
}

// ---------------------------------------------------------------------------
// Time-embedding MLP (1 x D work, one block): bias2[s] = act_b + te(s)
// ---------------------------------------------------------------------------
__global__ __launch_bounds__(512) void te_kernel(const float* __restrict__ w0, const float* __restrict__ b0,
                                                 const float* __restrict__ w1, const float* __restrict__ b1,
                                                 const float* __restrict__ w2, const float* __restrict__ b2,
                                                 const float* __restrict__ act_b,
                                                 float* __restrict__ bias2) {
  __shared__ float sh0[128];
  __shared__ float sh1[256];
  int j = threadIdx.x;
  for (int s = 0; s < NSTEPS; ++s) {
    float t = (float)s * (1.f / (float)NSTEPS);
    if (j < 128) sh0[j] = dev_silu(t * w0[j] + b0[j]);
    __syncthreads();
    if (j < 256) {
      float a = b1[j];
      for (int k = 0; k < 128; ++k) a += sh0[k] * w1[k * 256 + j];
      sh1[j] = dev_silu(a);
    }
    __syncthreads();
    {
      float a = b2[j];
      for (int k = 0; k < 256; ++k) a += sh1[k] * w2[k * 512 + j];
      bias2[s * DMODEL + j] = act_b[j] + a;
    }
    __syncthreads();
  }
}

// ---------------------------------------------------------------------------
// Core bf16 WMMA GEMM:  C[M,N] = act(A[M,K] @ B[K,N] + bias)
// Block 256 thr = 8 wave32 (4M x 2N), wave tile 32x64, block tile 128x128.
// A row-major bf16; B pre-packed fragment order; f32 accumulate.
// ---------------------------------------------------------------------------
template <bool SILU, bool GELU, bool WF32, bool WBF16>
__global__ __launch_bounds__(256) void gemm_bf16_wmma(const __bf16* __restrict__ A,
                                                      const __bf16* __restrict__ Bp,
                                                      const float* __restrict__ bias,
                                                      float* __restrict__ Cf,
                                                      __bf16* __restrict__ Cb,
                                                      int N, int K) {
  const int lane  = threadIdx.x & 31;
  const int wave  = threadIdx.x >> 5;
  const int waveM = wave >> 1;
  const int waveN = wave & 1;
  const int rowBase = blockIdx.x * 128 + waveM * 32;
  const int colBase = blockIdx.y * 128 + waveN * 64;
  if (colBase >= N) return;  // supports N=64 heads
  const int ktiles   = K >> 5;
  const int laneHalf = lane >> 4;   // 0 | 1
  const int lmod     = lane & 15;

  v8f acc[2][4];
  #pragma unroll
  for (int mi = 0; mi < 2; ++mi)
    #pragma unroll
    for (int ni = 0; ni < 4; ++ni)
      #pragma unroll
      for (int e = 0; e < 8; ++e) acc[mi][ni][e] = 0.f;

  const __bf16* a0 = A + (size_t)(rowBase + lmod) * K + laneHalf * 16;
  const __bf16* a1 = a0 + (size_t)16 * K;
  const __bf16* bbase = Bp + ((size_t)(colBase >> 4) * ktiles) * 512 + lane * 16;

  for (int kt = 0; kt < ktiles; ++kt) {
    FragAB a[2], b[4];
    const __bf16* pa0 = a0 + kt * 32;
    a[0].h[0] = *(const v8bf*)(pa0);
    a[0].h[1] = *(const v8bf*)(pa0 + 8);
    const __bf16* pa1 = a1 + kt * 32;
    a[1].h[0] = *(const v8bf*)(pa1);
    a[1].h[1] = *(const v8bf*)(pa1 + 8);
    #pragma unroll
    for (int ni = 0; ni < 4; ++ni) {
      const __bf16* pb = bbase + ((size_t)ni * ktiles + kt) * 512;
      b[ni].h[0] = *(const v8bf*)(pb);
      b[ni].h[1] = *(const v8bf*)(pb + 8);
    }
    // Unconditional speculative prefetch 4 k-tiles (4KB) ahead of the B
    // stream (global_prefetch_b8; past-end prefetches are dropped by HW,
    // keeps the hot loop branch-free).
    __builtin_prefetch(bbase + (size_t)(kt + 4) * 512, 0, 1);
    #pragma unroll
    for (int mi = 0; mi < 2; ++mi)
      #pragma unroll
      for (int ni = 0; ni < 4; ++ni)
        acc[mi][ni] = __builtin_amdgcn_wmma_f32_16x16x32_bf16(
            false, a[mi].v, false, b[ni].v, (short)0, acc[mi][ni], false, false);
  }

  #pragma unroll
  for (int ni = 0; ni < 4; ++ni) {
    int col = colBase + ni * 16 + lmod;
    float bv = bias[col];
    #pragma unroll
    for (int mi = 0; mi < 2; ++mi) {
      #pragma unroll
      for (int r = 0; r < 8; ++r) {
        int row = rowBase + mi * 16 + r + laneHalf * 8;
        float v = acc[mi][ni][r] + bv;
        if (SILU) v = dev_silu(v);
        if (GELU) v = dev_gelu(v);
        size_t idx = (size_t)row * N + col;
        if (WF32)  Cf[idx] = v;
        if (WBF16) Cb[idx] = (__bf16)v;
      }
    }
  }
}

// ---------------------------------------------------------------------------
// LayerNorm(h + res) over D=512, one wave32 per row (16 elems/lane).
// Writes both f32 (residual path) and bf16 (next-GEMM A operand).
// ---------------------------------------------------------------------------
__global__ __launch_bounds__(256) void ln_residual_kernel(const float* __restrict__ hin,
                                                          const float* __restrict__ res,
                                                          const float* __restrict__ g,
                                                          const float* __restrict__ bta,
                                                          float* __restrict__ hf,
                                                          __bf16* __restrict__ hb) {
  const int lane = threadIdx.x & 31;
  const int row  = blockIdx.x * 8 + (threadIdx.x >> 5);
  const size_t base = (size_t)row * DMODEL + lane * 16;
  float x[16];
  float s = 0.f, ss = 0.f;
  #pragma unroll
  for (int c = 0; c < 4; ++c) {
    float4 a = *(const float4*)(hin + base + c * 4);
    float4 b = *(const float4*)(res + base + c * 4);
    x[c * 4 + 0] = a.x + b.x; x[c * 4 + 1] = a.y + b.y;
    x[c * 4 + 2] = a.z + b.z; x[c * 4 + 3] = a.w + b.w;
  }
  #pragma unroll
  for (int i = 0; i < 16; ++i) { s += x[i]; ss += x[i] * x[i]; }
  #pragma unroll
  for (int m = 16; m >= 1; m >>= 1) {
    s  += __shfl_xor(s, m, 32);
    ss += __shfl_xor(ss, m, 32);
  }
  float mean = s * (1.f / (float)DMODEL);
  float var  = ss * (1.f / (float)DMODEL) - mean * mean;
  float rstd = rsqrtf(var + 1e-5f);
  #pragma unroll
  for (int i = 0; i < 16; ++i) {
    int col = lane * 16 + i;
    float y = (x[i] - mean) * rstd * g[col] + bta[col];
    hf[base + i] = y;
    hb[base + i] = (__bf16)y;
  }
}

// ---------------------------------------------------------------------------
// rsample + log-prob accumulation + flow update, one wave32 per row.
// (v - mean)/std == eps exactly, so lp term = -0.5*eps^2 - ls - 0.5*log(2pi).
// ---------------------------------------------------------------------------
__global__ __launch_bounds__(256) void sample_kernel(const float* __restrict__ mean,
                                                     const float* __restrict__ lsr,
                                                     const float* __restrict__ eps,
                                                     float* __restrict__ xf,
                                                     __bf16* __restrict__ xb,
                                                     float* __restrict__ lp) {
  const int lane = threadIdx.x & 31;
  const int row  = blockIdx.x * 8 + (threadIdx.x >> 5);
  const float log2pi = 1.8378770664093453f;
  float acc = 0.f;
  #pragma unroll
  for (int t = 0; t < 2; ++t) {
    int j = lane + t * 32;
    size_t i = (size_t)row * ADIM + j;
    float ls = tanhf(lsr[i]);
    ls = -5.f + 3.5f * (ls + 1.f);          // LOG_STD_MIN + 0.5*(MAX-MIN)*(ls+1)
    float st = __expf(ls);
    float e  = eps[i];
    float v  = mean[i] + st * e;
    acc += -0.5f * e * e - ls - 0.5f * log2pi;
    float xn = xf[i] + 0.25f * v;           // x += v / STEPS
    xf[i] = xn;
    xb[i] = (__bf16)xn;
  }
  #pragma unroll
  for (int m = 16; m >= 1; m >>= 1) acc += __shfl_xor(acc, m, 32);
  if (lane == 0) lp[row] += acc;
}

__global__ __launch_bounds__(256) void final_kernel(const float* __restrict__ xf,
                                                    const float* __restrict__ scale,
                                                    const float* __restrict__ bias,
                                                    const float* __restrict__ lp,
                                                    float* __restrict__ out) {
  const int lane = threadIdx.x & 31;
  const int row  = blockIdx.x * 8 + (threadIdx.x >> 5);
  float acc = 0.f;
  #pragma unroll
  for (int t = 0; t < 2; ++t) {
    int j = lane + t * 32;
    size_t i = (size_t)row * ADIM + j;
    float y = tanhf(xf[i]);
    out[i] = y * scale[j] + bias[j];
    acc += logf(scale[j] * (1.f - y * y) + 1e-6f);
  }
  #pragma unroll
  for (int m = 16; m >= 1; m >>= 1) acc += __shfl_xor(acc, m, 32);
  if (lane == 0) out[(size_t)B_ROWS * ADIM + row] = lp[row] - acc;
}

// ---------------------------------------------------------------------------
// Host orchestration
// ---------------------------------------------------------------------------
enum GemmMode { GF32 = 0, GBF16 = 1, GSILU_BF16 = 2, GGELU_BF16 = 3, GBOTH = 4 };

static void run_gemm(int mode, const __bf16* A, const __bf16* Bp, const float* bias,
                     float* Cf, __bf16* Cb, int N, int K, hipStream_t st) {
  dim3 grid(B_ROWS / 128, (N + 127) / 128), blk(256);
  switch (mode) {
    case GF32:       gemm_bf16_wmma<false, false, true,  false><<<grid, blk, 0, st>>>(A, Bp, bias, Cf, Cb, N, K); break;
    case GBF16:      gemm_bf16_wmma<false, false, false, true ><<<grid, blk, 0, st>>>(A, Bp, bias, Cf, Cb, N, K); break;
    case GSILU_BF16: gemm_bf16_wmma<true,  false, false, true ><<<grid, blk, 0, st>>>(A, Bp, bias, Cf, Cb, N, K); break;
    case GGELU_BF16: gemm_bf16_wmma<false, true,  false, true ><<<grid, blk, 0, st>>>(A, Bp, bias, Cf, Cb, N, K); break;
    case GBOTH:      gemm_bf16_wmma<false, false, true,  true ><<<grid, blk, 0, st>>>(A, Bp, bias, Cf, Cb, N, K); break;
  }
}

extern "C" void kernel_launch(void* const* d_in, const int* in_sizes, int n_in,
                              void* d_out, int out_size, void* d_ws, size_t ws_size,
                              hipStream_t stream) {
  // Inputs (setup_inputs dict order, params flattened in insertion order)
  const float* obs    = (const float*)d_in[0];
  const float* x0     = (const float*)d_in[1];
  const float* eps    = (const float*)d_in[2];
  const float* enc_w0 = (const float*)d_in[3];
  const float* enc_b0 = (const float*)d_in[4];
  const float* enc_w1 = (const float*)d_in[5];
  const float* enc_b1 = (const float*)d_in[6];
  const float* act_w  = (const float*)d_in[7];
  const float* act_b  = (const float*)d_in[8];
  const float* t_w0   = (const float*)d_in[9];
  const float* t_b0   = (const float*)d_in[10];
  const float* t_w1   = (const float*)d_in[11];
  const float* t_b1   = (const float*)d_in[12];
  const float* t_w2   = (const float*)d_in[13];
  const float* t_b2   = (const float*)d_in[14];
  const float* sa_wv  = (const float*)d_in[15];
  const float* sa_bv  = (const float*)d_in[16];
  const float* sa_wo  = (const float*)d_in[17];
  const float* sa_bo  = (const float*)d_in[18];
  const float* ca_wv  = (const float*)d_in[19];
  const float* ca_bv  = (const float*)d_in[20];
  const float* ca_wo  = (const float*)d_in[21];
  const float* ca_bo  = (const float*)d_in[22];
  const float* ff_w1  = (const float*)d_in[23];
  const float* ff_b1  = (const float*)d_in[24];
  const float* ff_w2  = (const float*)d_in[25];
  const float* ff_b2  = (const float*)d_in[26];
  const float* ln1_g  = (const float*)d_in[27];
  const float* ln1_b  = (const float*)d_in[28];
  const float* ln2_g  = (const float*)d_in[29];
  const float* ln2_b  = (const float*)d_in[30];
  const float* ln3_g  = (const float*)d_in[31];
  const float* ln3_b  = (const float*)d_in[32];
  const float* vm_w   = (const float*)d_in[33];
  const float* vm_b   = (const float*)d_in[34];
  const float* vs_w   = (const float*)d_in[35];
  const float* vs_b   = (const float*)d_in[36];
  const float* scale  = (const float*)d_in[37];
  const float* bias   = (const float*)d_in[38];

  char* wsb = (char*)d_ws;
  size_t off = 0;
  auto alloc = [&](size_t bytes) -> char* {
    char* p = wsb + off;
    off += (bytes + 255) & ~(size_t)255;
    return p;
  };

  __bf16* obs_bf  = (__bf16*)alloc((size_t)B_ROWS * 1024 * 2);
  __bf16* xb      = (__bf16*)alloc((size_t)B_ROWS * ADIM * 2);
  float*  xf      = (float*) alloc((size_t)B_ROWS * ADIM * 4);
  float*  lp      = (float*) alloc((size_t)B_ROWS * 4);
  float*  bias2   = (float*) alloc((size_t)NSTEPS * DMODEL * 4);
  __bf16* hb      = (__bf16*)alloc((size_t)B_ROWS * DMODEL * 2);
  float*  hf      = (float*) alloc((size_t)B_ROWS * DMODEL * 4);
  float*  resf    = (float*) alloc((size_t)B_ROWS * DMODEL * 4);
  __bf16* tmpbf   = (__bf16*)alloc((size_t)B_ROWS * 2048 * 2);   // ff1 out / v-proj outs / enc hidden
  float*  caf     = (float*) alloc((size_t)NLAYER * B_ROWS * DMODEL * 4);
  float*  meanf   = (float*) alloc((size_t)B_ROWS * ADIM * 4);
  float*  lsf     = (float*) alloc((size_t)B_ROWS * ADIM * 4);
  __bf16* obs_emb = (__bf16*)alloc((size_t)B_ROWS * DMODEL * 2);

  auto packW = [&](const float* src, int K, int N) -> __bf16* {
    __bf16* dst = (__bf16*)alloc((size_t)K * N * 2);
    int total = K * N;
    pack_b_kernel<<<(total + 255) / 256, 256, 0, stream>>>(src, dst, K, N);
    return dst;
  };
  auto packWL = [&](const float* src, int L, int K, int N) -> __bf16* {
    __bf16* dst = (__bf16*)alloc((size_t)L * K * N * 2);
    int total = K * N;
    for (int l = 0; l < L; ++l)
      pack_b_kernel<<<(total + 255) / 256, 256, 0, stream>>>(
          src + (size_t)l * K * N, dst + (size_t)l * K * N, K, N);
    return dst;
  };

  __bf16* enc_w0p = packW(enc_w0, 1024, 256);
  __bf16* enc_w1p = packW(enc_w1, 256, DMODEL);
  __bf16* act_wp  = packW(act_w, ADIM, DMODEL);
  __bf16* sa_wv_p = packWL(sa_wv, NLAYER, DMODEL, DMODEL);
  __bf16* sa_wo_p = packWL(sa_wo, NLAYER, DMODEL, DMODEL);
  __bf16* ca_wv_p = packWL(ca_wv, NLAYER, DMODEL, DMODEL);
  __bf16* ca_wo_p = packWL(ca_wo, NLAYER, DMODEL, DMODEL);
  __bf16* ff_w1p  = packWL(ff_w1, NLAYER, DMODEL, 2048);
  __bf16* ff_w2p  = packWL(ff_w2, NLAYER, 2048, DMODEL);
  __bf16* vm_wp   = packW(vm_w, DMODEL, ADIM);
  __bf16* vs_wp   = packW(vs_w, DMODEL, ADIM);

  // Init / tiny precompute
  {
    int n = B_ROWS * 1024;
    f32_to_bf16_kernel<<<(n + 255) / 256, 256, 0, stream>>>(obs, obs_bf, n);
  }
  {
    int n = B_ROWS * ADIM;
    x_init_kernel<<<(n + 255) / 256, 256, 0, stream>>>(x0, xf, xb, n);
  }
  lp_init_kernel<<<(B_ROWS + 255) / 256, 256, 0, stream>>>(x0, lp);
  te_kernel<<<1, 512, 0, stream>>>(t_w0, t_b0, t_w1, t_b1, t_w2, t_b2, act_b, bias2);

  // Encoder: obs_emb = silu(obs@W0+b0)@W1+b1
  run_gemm(GSILU_BF16, obs_bf, enc_w0p, enc_b0, nullptr, tmpbf, 256, 1024, stream);
  run_gemm(GBF16, tmpbf, enc_w1p, enc_b1, nullptr, obs_emb, DMODEL, 256, stream);

  // Cross-attention is step-invariant: hoist out of the denoising loop.
  for (int l = 0; l < NLAYER; ++l) {
    run_gemm(GBF16, obs_emb, ca_wv_p + (size_t)l * DMODEL * DMODEL, ca_bv + l * DMODEL,
             nullptr, tmpbf, DMODEL, DMODEL, stream);
    run_gemm(GF32, tmpbf, ca_wo_p + (size_t)l * DMODEL * DMODEL, ca_bo + l * DMODEL,
             caf + (size_t)l * B_ROWS * DMODEL, nullptr, DMODEL, DMODEL, stream);
  }

  dim3 rgrid(B_ROWS / 8), rblk(256);
  for (int s = 0; s < NSTEPS; ++s) {
    // h = x@act_w + (act_b + te[s])
    run_gemm(GBOTH, xb, act_wp, bias2 + s * DMODEL, hf, hb, DMODEL, ADIM, stream);
    for (int l = 0; l < NLAYER; ++l) {
      // self-attn (seq_len==1 => out_proj(v_proj(h)))
      run_gemm(GBF16, hb, sa_wv_p + (size_t)l * DMODEL * DMODEL, sa_bv + l * DMODEL,
               nullptr, tmpbf, DMODEL, DMODEL, stream);
      run_gemm(GF32, tmpbf, sa_wo_p + (size_t)l * DMODEL * DMODEL, sa_bo + l * DMODEL,
               resf, nullptr, DMODEL, DMODEL, stream);
      ln_residual_kernel<<<rgrid, rblk, 0, stream>>>(hf, resf, ln1_g + l * DMODEL, ln1_b + l * DMODEL, hf, hb);
      // cross-attn residual (precomputed)
      ln_residual_kernel<<<rgrid, rblk, 0, stream>>>(hf, caf + (size_t)l * B_ROWS * DMODEL,
                                                     ln2_g + l * DMODEL, ln2_b + l * DMODEL, hf, hb);
      // feed-forward
      run_gemm(GGELU_BF16, hb, ff_w1p + (size_t)l * DMODEL * 2048, ff_b1 + l * 2048,
               nullptr, tmpbf, 2048, DMODEL, stream);
      run_gemm(GF32, tmpbf, ff_w2p + (size_t)l * 2048 * DMODEL, ff_b2 + l * DMODEL,
               resf, nullptr, DMODEL, 2048, stream);
      ln_residual_kernel<<<rgrid, rblk, 0, stream>>>(hf, resf, ln3_g + l * DMODEL, ln3_b + l * DMODEL, hf, hb);
    }
    // heads + rsample + flow update
    run_gemm(GF32, hb, vm_wp, vm_b, meanf, nullptr, ADIM, DMODEL, stream);
    run_gemm(GF32, hb, vs_wp, vs_b, lsf, nullptr, ADIM, DMODEL, stream);
    sample_kernel<<<rgrid, rblk, 0, stream>>>(meanf, lsf, eps + (size_t)s * B_ROWS * ADIM, xf, xb, lp);
  }

  final_kernel<<<rgrid, rblk, 0, stream>>>(xf, scale, bias, lp, (float*)d_out);
}
